// ConduitBattery_10986526343844
// MI455X (gfx1250) — matrix operand order repaired and checked by reading, not compiled, not run on hardware
//
#include <hip/hip_runtime.h>
#include <hip/hip_bf16.h>
#include <math.h>
#include <string.h>

// ---------------------------------------------------------------------------
// MI455X (gfx1250, wave32) implementation.
// All dense linear algebra runs through v_wmma_f32_16x16x32_bf16 (f32 acc).
// One wave computes a 16x64 D tile (4 independent accumulators -> 4 dependent-
// free WMMAs per K chunk). Steady-state loop: b128 clause loads + prefetch +
// 4 WMMAs; the partial-K chunk is peeled out of the loop (register masking).
// ---------------------------------------------------------------------------

typedef __attribute__((ext_vector_type(16))) __bf16         v16bf;
typedef __attribute__((ext_vector_type(16))) unsigned short v16u;
typedef __attribute__((ext_vector_type(8)))  unsigned short u16x8;
typedef __attribute__((ext_vector_type(8)))  float          v8f;

#define NT 4   // N tiles per wave

static __device__ __forceinline__ unsigned short f2bfu(float f) {
  unsigned u = __builtin_bit_cast(unsigned, f);
  unsigned r = u + 0x7FFFu + ((u >> 16) & 1u);   // round-to-nearest-even
  return (unsigned short)(r >> 16);
}

// ---------------------------------------------------------------------------
// Generic batched WMMA GEMM:  C[b,h] = act(alpha * A @ B^T + bias) (+ resid)
//   A : bf16 [M,K], row stride lda (lda % 8 == 0), batch offsets sAb/sAh
//   B : bf16 [N,K] row stride ldb (ldb % 8 == 0), zero-padded to K rounded up
//   act: 0 none, 1 exact GELU, 2 tanh
// ---------------------------------------------------------------------------
struct GemmP {
  const unsigned short* A;
  const unsigned short* B;
  const float* bias;
  const float* resid;
  const float* alpha_ptr;   // optional device scalar multiplier (e.g. "gain")
  float* Cf;
  unsigned short* Cb;
  long sAb, sAh, sBb, sBh, sCb, sCh, sRb, sRh;
  int M, N, K, lda, ldb, ldc, H, act;
  float alpha;
};

__global__ __launch_bounds__(32) void k_gemm(GemmP p) {
  int lane = threadIdx.x;
  int b = blockIdx.z / p.H, h = blockIdx.z % p.H;
  const unsigned short* A = p.A + (long)b * p.sAb + (long)h * p.sAh;
  const unsigned short* B = p.B + (long)b * p.sBb + (long)h * p.sBh;

  int m0 = blockIdx.y * 16, n0 = blockIdx.x * (16 * NT);
  int mq = m0 + (lane & 15); mq = (mq < p.M) ? mq : (p.M - 1);        // clamp
  int kb = (lane >> 4) * 16;     // A-fragment K sub-base (kb+e)
  int kB = (lane >> 4) * 16;     // B-fragment K sub-base (kB+e)
  const unsigned short* Arow = A + (long)mq * p.lda;

  int nr[NT];
  const unsigned short* Brow[NT];
  v8f acc[NT];
#pragma unroll
  for (int t = 0; t < NT; ++t) {
    nr[t] = n0 + t * 16 + (lane & 15);                                // real n
    int nq = (nr[t] < p.N) ? nr[t] : (p.N - 1);                       // clamp
    Brow[t] = B + (long)nq * p.ldb;
    acc[t] = (v8f)0.0f;
  }

  int kmain = p.K & ~31;                      // full 32-wide chunks
  for (int k0 = 0; k0 < kmain; k0 += 32) {
    // speculative prefetch of next chunk (ISA: dropped on translation fail)
    __builtin_prefetch(Arow + k0 + 32 + kb, 0, 1);
    __builtin_prefetch(Brow[0] + k0 + 32 + kB, 0, 1);
    u16x8 a0 = *(const u16x8*)(Arow + k0 + kb);
    u16x8 a1 = *(const u16x8*)(Arow + k0 + kb + 8);
    v16bf av = __builtin_bit_cast(v16bf,
        __builtin_shufflevector(a0, a1, 0,1,2,3,4,5,6,7,8,9,10,11,12,13,14,15));
#pragma unroll
    for (int t = 0; t < NT; ++t) {
      u16x8 b0 = *(const u16x8*)(Brow[t] + k0 + kB);
      u16x8 b1 = *(const u16x8*)(Brow[t] + k0 + kB + 8);
      v16bf bv = __builtin_bit_cast(v16bf,
          __builtin_shufflevector(b0, b1, 0,1,2,3,4,5,6,7,8,9,10,11,12,13,14,15));
      acc[t] = __builtin_amdgcn_wmma_f32_16x16x32_bf16(
          false, av, false, bv, (short)0, acc[t], false, false);
    }
  }
  if (kmain < p.K) {                          // peeled tail chunk (K % 32 != 0)
    int k0 = kmain;
    u16x8 a0 = *(const u16x8*)(Arow + k0 + kb);
    u16x8 a1 = *(const u16x8*)(Arow + k0 + kb + 8);
    v16u ua = __builtin_shufflevector(a0, a1, 0,1,2,3,4,5,6,7,8,9,10,11,12,13,14,15);
#pragma unroll
    for (int e = 0; e < 16; ++e)
      if (k0 + kb + e >= p.K) ua[e] = 0;
    v16bf av = __builtin_bit_cast(v16bf, ua);
#pragma unroll
    for (int t = 0; t < NT; ++t) {
      u16x8 b0 = *(const u16x8*)(Brow[t] + k0 + kB);
      u16x8 b1 = *(const u16x8*)(Brow[t] + k0 + kB + 8);
      v16u ub = __builtin_shufflevector(b0, b1, 0,1,2,3,4,5,6,7,8,9,10,11,12,13,14,15);
#pragma unroll
      for (int e = 0; e < 16; ++e)
        if (k0 + kB + e >= p.K) ub[e] = 0;
      acc[t] = __builtin_amdgcn_wmma_f32_16x16x32_bf16(
          false, av, false, __builtin_bit_cast(v16bf, ub),
          (short)0, acc[t], false, false);
    }
  }

  float al = p.alpha;
  if (p.alpha_ptr) al *= *p.alpha_ptr;
  int mbase = m0 + ((lane >> 4) << 3);
#pragma unroll
  for (int t = 0; t < NT; ++t) {
    int n = nr[t];
#pragma unroll
    for (int r2 = 0; r2 < 8; ++r2) {  // D layout: VGPR r -> row r + 8*(lane>=16)
      int m = mbase + r2;
      if (m < p.M) {
        long co = (long)b * p.sCb + (long)h * p.sCh + (long)m * p.ldc + n;
        if (n < p.N) {
          float v = acc[t][r2] * al;
          if (p.bias) v += p.bias[n];
          if (p.act == 1)      v = 0.5f * v * (1.f + erff(v * 0.70710678118f));
          else if (p.act == 2) v = tanhf(v);
          if (p.resid) v += p.resid[(long)b * p.sRb + (long)h * p.sRh + (long)m * p.ldc + n];
          if (p.Cf) p.Cf[co] = v;
          if (p.Cb) p.Cb[co] = f2bfu(v);
        }
      }
    }
  }
}

// Weight f32 [N,K] -> bf16 [N,Kp] zero-padded (Kp = round_up(K,32))
__global__ void k_wcvt(const float* __restrict__ w, unsigned short* __restrict__ o,
                       int N, int K, int Kp) {
  long i = (long)blockIdx.x * blockDim.x + threadIdx.x;
  if (i >= (long)N * Kp) return;
  long row = i / Kp; int col = (int)(i % Kp);
  o[i] = (col < K) ? f2bfu(w[row * (long)K + col]) : (unsigned short)0;
}

// Vt[(b*H+h)*dh + d][k] = qkv[b][k][2D + h*dh + d]   (V slice transpose, bf16)
__global__ void k_vtrans(const unsigned short* __restrict__ qkv,
                         unsigned short* __restrict__ vt,
                         int D, int dh, int H, long total) {
  long i = (long)blockIdx.x * blockDim.x + threadIdx.x;
  if (i >= total) return;
  int k = (int)(i & 127);
  long r = i >> 7;            // (b*H+h)*dh + d
  int d = (int)(r % dh);
  long bh = r / dh;
  int b = (int)(bh / H), h = (int)(bh % H);
  vt[i] = qkv[(long)b * 128 * 3 * D + (long)k * 3 * D + 2 * D + h * dh + d];
}

// ---------------------------------------------------------------------------
// LayerNorm over last dim D (f32 in -> bf16 out, row stride ldout, pads zeroed)
// ---------------------------------------------------------------------------
__global__ __launch_bounds__(256) void k_ln(const float* __restrict__ x,
                                            const float* __restrict__ g,
                                            const float* __restrict__ b,
                                            unsigned short* __restrict__ out,
                                            int D, int ldout) {
  __shared__ float red[256];
  __shared__ float stat[2];
  long row = blockIdx.x;
  const float* xr = x + row * (long)D;
  int tid = threadIdx.x;
  float s = 0.f;
  for (int j = tid; j < D; j += 256) s += xr[j];
  red[tid] = s; __syncthreads();
  for (int o = 128; o > 0; o >>= 1) { if (tid < o) red[tid] += red[tid + o]; __syncthreads(); }
  if (tid == 0) stat[0] = red[0] / D;
  __syncthreads();
  float mean = stat[0];
  float s2 = 0.f;
  for (int j = tid; j < D; j += 256) { float d = xr[j] - mean; s2 += d * d; }
  red[tid] = s2; __syncthreads();
  for (int o = 128; o > 0; o >>= 1) { if (tid < o) red[tid] += red[tid + o]; __syncthreads(); }
  if (tid == 0) stat[1] = rsqrtf(red[0] / D + 1e-5f);
  __syncthreads();
  float rstd = stat[1];
  unsigned short* orow = out + row * (long)ldout;
  for (int j = tid; j < D; j += 256)
    orow[j] = f2bfu((xr[j] - mean) * rstd * g[j] + b[j]);
  for (int j = D + tid; j < ldout; j += 256) orow[j] = 0;
}

// Softmax over rows of 128 (f32 -> bf16)
__global__ __launch_bounds__(128) void k_softmax(const float* __restrict__ s,
                                                 unsigned short* __restrict__ o) {
  __shared__ float red[128];
  long row = blockIdx.x;
  int tid = threadIdx.x;
  float v = s[row * 128 + tid];
  red[tid] = v; __syncthreads();
  for (int off = 64; off > 0; off >>= 1) { if (tid < off) red[tid] = fmaxf(red[tid], red[tid + off]); __syncthreads(); }
  float mx = red[0]; __syncthreads();
  float e = expf(v - mx);
  red[tid] = e; __syncthreads();
  for (int off = 64; off > 0; off >>= 1) { if (tid < off) red[tid] += red[tid + off]; __syncthreads(); }
  o[row * 128 + tid] = f2bfu(e / red[0]);
}

// r2 = r + depthwise_conv3(r) over N dim (per batch of 4096 tokens, RANK=16)
__global__ void k_dwconv(const float* __restrict__ r, const float* __restrict__ w,
                         float* __restrict__ out, long total) {
  long idx = (long)blockIdx.x * blockDim.x + threadIdx.x;
  if (idx >= total) return;
  int c = (int)(idx & 15);
  long t = idx >> 4;
  int n = (int)(t & 4095);
  float x0 = r[idx];
  float xm = (n > 0)    ? r[idx - 16] : 0.f;
  float xp = (n < 4095) ? r[idx + 16] : 0.f;
  out[idx] = x0 + w[c * 3 + 0] * xm + w[c * 3 + 1] * x0 + w[c * 3 + 2] * xp;
}

// ---------------------------------------------------------------------------
// Per-group thin SVD (32x16) via Jacobi on Gram matrix + spectral features.
// One wave per group; single-wave LDS ops are hardware-ordered (DScnt).
// ---------------------------------------------------------------------------
__global__ __launch_bounds__(32) void k_svd(const float* __restrict__ r,
                                            const float* __restrict__ ema,
                                            float* __restrict__ U, float* __restrict__ Vh,
                                            float* __restrict__ S, float* __restrict__ rowscale,
                                            float* __restrict__ gfeats, float* __restrict__ cfeats) {
  __shared__ float W[32][16];
  __shared__ float Gm[16][16];
  __shared__ float V[16][16];
  __shared__ float VhS[16][16];
  __shared__ float Ush[32][16];
  __shared__ float lam[16];
  __shared__ int   perm[16];
  __shared__ float Sv[16];
  __shared__ float rsS[32];
  long g = blockIdx.x;
  int lane = threadIdx.x;

  const float* row = r + (g * 32 + lane) * 16;
  float v[16]; float nrm = 0.f;
#pragma unroll
  for (int c = 0; c < 16; ++c) { v[c] = row[c]; nrm += v[c] * v[c]; }
  float rs = fmaxf(sqrtf(nrm), 1e-6f);
  rsS[lane] = rs;
  rowscale[g * 32 + lane] = rs;
#pragma unroll
  for (int c = 0; c < 16; ++c) W[lane][c] = v[c] / rs;
  __syncthreads();

  for (int t = 0; t < 8; ++t) {       // Gram = W^T W, V = I
    int idx = lane * 8 + t; int i = idx >> 4, j = idx & 15;
    float s = 0.f;
    for (int rr = 0; rr < 32; ++rr) s += W[rr][i] * W[rr][j];
    Gm[i][j] = s;
    V[i][j] = (i == j) ? 1.f : 0.f;
  }
  __syncthreads();

  for (int sweep = 0; sweep < 10; ++sweep)
    for (int p = 0; p < 15; ++p)
      for (int q = p + 1; q < 16; ++q) {
        float apq = Gm[p][q];
        if (fabsf(apq) > 1e-12f) {            // uniform branch (LDS scalar)
          float app = Gm[p][p], aqq = Gm[q][q];
          float tau = (aqq - app) / (2.f * apq);
          float t = ((tau >= 0.f) ? 1.f : -1.f) / (fabsf(tau) + sqrtf(1.f + tau * tau));
          float c = rsqrtf(1.f + t * t), sn = t * c;
          int j = lane & 15;
          if (lane < 16) {                    // rows p,q  (J^T A)
            float gp = Gm[p][j], gq = Gm[q][j];
            Gm[p][j] = c * gp - sn * gq;
            Gm[q][j] = sn * gp + c * gq;
          }
          if (lane < 16) {                    // cols p,q  (... J) and V <- V J
            float gip = Gm[j][p], giq = Gm[j][q];
            Gm[j][p] = c * gip - sn * giq;
            Gm[j][q] = sn * gip + c * giq;
            float vip = V[j][p], viq = V[j][q];
            V[j][p] = c * vip - sn * viq;
            V[j][q] = sn * vip + c * viq;
          }
        }
      }
  __syncthreads();

  if (lane < 16) lam[lane] = Gm[lane][lane];
  __syncthreads();
  if (lane < 16) {                            // parallel rank sort (descending)
    float mi = lam[lane]; int rank = 0;
    for (int j = 0; j < 16; ++j) {
      float lj = lam[j];
      if (lj > mi || (lj == mi && j < lane)) rank++;
    }
    perm[rank] = lane;
  }
  __syncthreads();
  if (lane < 16) {
    int src = perm[lane];
    Sv[lane] = sqrtf(fmaxf(lam[src], 0.f));
    for (int j = 0; j < 16; ++j) VhS[lane][j] = V[j][src];
  }
  __syncthreads();

#pragma unroll
  for (int k = 0; k < 16; ++k) {              // U = W V S^-1  (lane = row)
    float s = Sv[k];
    float inv = (s > 1e-8f) ? 1.f / s : 0.f;
    float d = 0.f;
#pragma unroll
    for (int c = 0; c < 16; ++c) d += W[lane][c] * VhS[k][c];
    float u = d * inv;
    Ush[lane][k] = u;
    U[(g * 32 + lane) * 16 + k] = u;
  }
  __syncthreads();

  const float eps = 1e-8f;
  float ssum = 0.f, smax = -1e30f, smin = 1e30f;
  for (int k = 0; k < 16; ++k) { float s = Sv[k]; ssum += s; smax = fmaxf(smax, s); smin = fminf(smin, s); }
  float ent = 0.f, nov = 0.f;
  for (int k = 0; k < 16; ++k) {
    float pk = Sv[k] / (ssum + eps);
    ent -= pk * logf(pk + eps);
    nov += fabsf(pk - ema[k]);
  }
  float eff = expf(ent);
  float rsm = 0.f;
  for (int rr = 0; rr < 32; ++rr) rsm += rsS[rr];
  rsm *= (1.f / 32.f);

  if (lane < 16) {
    int k = lane;
    float s = Sv[k];
    float pk = s / (ssum + eps);
    float cum = 0.f;
    for (int j = 0; j <= k; ++j) cum += Sv[j] / (ssum + eps);
    float ue = 0.f; for (int rr = 0; rr < 32; ++rr) { float u = Ush[rr][k]; ue += u * u; } ue *= (1.f / 32.f);
    float ve = 0.f; for (int j = 0; j < 16; ++j) { float w = VhS[k][j]; ve += w * w; } ve *= (1.f / 16.f);
    float* gf = gfeats + g * 119;
    gf[k] = s; gf[16 + k] = pk; gf[32 + k] = logf(s + eps); gf[48 + k] = cum;
    gf[64 + k] = sqrtf(pk + eps); gf[80 + k] = ue; gf[96 + k] = ve;
    float* cf = cfeats + g * 51;
    cf[k] = s; cf[16 + k] = pk; cf[32 + k] = VhS[0][k];
    S[g * 16 + k] = s;
    for (int j = 0; j < 16; ++j) Vh[(g * 16 + k) * 16 + j] = VhS[k][j];
  }
  if (lane == 0) {
    float* gf = gfeats + g * 119;
    gf[112] = ent; gf[113] = eff; gf[114] = nov; gf[115] = ssum;
    gf[116] = smax; gf[117] = smin; gf[118] = rsm;
    float* cf = cfeats + g * 51;
    cf[48] = ent; cf[49] = eff; cf[50] = nov;
  }
}

// recon = (U * S_sh) @ Vh * scale * row_scale + tanh(rf)  -> bf16 [B*N,32] (K-pad)
__global__ __launch_bounds__(32) void k_recon(const float* __restrict__ U,
                                              const float* __restrict__ Vh,
                                              const float* __restrict__ S,
                                              const float* __restrict__ rowscale,
                                              const float* __restrict__ delta,
                                              const float* __restrict__ alpha_logits,
                                              const float* __restrict__ scaleg,
                                              const float* __restrict__ rf,
                                              unsigned short* __restrict__ out) {
  __shared__ float ssh[16], vh[16][16], tr[16];
  long g = blockIdx.x; int lane = threadIdx.x;
  if (lane < 16) {
    float al = 0.2f / (1.f + expf(-alpha_logits[lane]));        // MAX_SPEC*sigmoid
    ssh[lane] = S[g * 16 + lane] * (1.f + al * delta[g * 16 + lane]);
    tr[lane] = tanhf(rf[g * 16 + lane]);
    for (int j = 0; j < 16; ++j) vh[lane][j] = Vh[(g * 16 + lane) * 16 + j];
  }
  __syncthreads();
  float sc = 1.f + 0.1f * scaleg[g];                            // 1 + MAX_SCALE*tanh
  float rs = rowscale[g * 32 + lane];
  float u[16];
#pragma unroll
  for (int k = 0; k < 16; ++k) u[k] = U[(g * 32 + lane) * 16 + k] * ssh[k];
  unsigned short* orow = out + (g * 32 + lane) * 32;
#pragma unroll
  for (int d = 0; d < 16; ++d) {
    float acc = 0.f;
#pragma unroll
    for (int k = 0; k < 16; ++k) acc += u[k] * vh[k][d];
    orow[d] = f2bfu(acc * sc * rs + tr[d]);
    orow[16 + d] = 0;
  }
}

// ----- small glue kernels --------------------------------------------------
__global__ void k_cvt(const float* x, unsigned short* o, long total) {
  long i = (long)blockIdx.x * blockDim.x + threadIdx.x;
  if (i < total) o[i] = f2bfu(x[i]);
}
__global__ void k_cmod(const float* film, const float* content, float* cmod,
                       float* batcat, long total) {
  long i = (long)blockIdx.x * blockDim.x + threadIdx.x;
  if (i >= total) return;
  long row = i / 768; int col = (int)(i % 768);
  float ga = film[row * 1536 + col], be = film[row * 1536 + 768 + col];
  float cm = content[i] * (1.f + ga) + be;
  cmod[i] = cm;
  batcat[row * 832 + col] = cm;
}
__global__ void k_batgeom(const float* geom, float* batcat, long total) {
  long i = (long)blockIdx.x * blockDim.x + threadIdx.x;
  if (i >= total) return;
  long row = i >> 6; int col = (int)(i & 63);
  batcat[row * 832 + 768 + col] = geom[i];
}
__global__ void k_rfcat(const float* S, const float* content, const float* geom,
                        float* rfcat, long total) {
  long i = (long)blockIdx.x * blockDim.x + threadIdx.x;
  if (i >= total) return;
  long row = i / 848; int col = (int)(i % 848);
  float v = (col < 16) ? S[row * 16 + col]
          : (col < 784) ? content[row * 768 + (col - 16)]
                        : geom[row * 64 + (col - 784)];
  rfcat[i] = v;
}
// gfeat = concat(mean_G(cmod), mean_G(battery), mean_G(geom), mean_G(S))
__global__ void k_gfeat(const float* cmod, const float* battery, const float* geom,
                        const float* S, float* gfeat) {
  int b = blockIdx.x;
  for (int j = threadIdx.x; j < 1616; j += blockDim.x) {
    float s = 0.f;
    if (j < 768)       { const float* p = cmod    + (long)b * 128 * 768 + j;          for (int g2 = 0; g2 < 128; ++g2) s += p[(long)g2 * 768]; }
    else if (j < 1536) { const float* p = battery + (long)b * 128 * 768 + (j - 768);  for (int g2 = 0; g2 < 128; ++g2) s += p[(long)g2 * 768]; }
    else if (j < 1600) { const float* p = geom    + (long)b * 128 * 64  + (j - 1536); for (int g2 = 0; g2 < 128; ++g2) s += p[(long)g2 * 64]; }
    else               { const float* p = S       + (long)b * 128 * 16  + (j - 1600); for (int g2 = 0; g2 < 128; ++g2) s += p[(long)g2 * 16]; }
    gfeat[b * 1616 + j] = s * (1.f / 128.f);
  }
}

// ---------------------------------------------------------------------------
// Host orchestration
// ---------------------------------------------------------------------------
static GemmP G0() { GemmP p; memset(&p, 0, sizeof(p)); p.H = 1; p.alpha = 1.f; return p; }
static void gemm_launch(hipStream_t st, GemmP p, int batch) {
  dim3 grid((p.N + 16 * NT - 1) / (16 * NT), (p.M + 15) / 16, batch);
  k_gemm<<<grid, dim3(32), 0, st>>>(p);
}

// Input leaf indices: setup_inputs() pytree flattened depth-first in dict
// insertion order (x, then params in declaration order) -> 91 leaves.
enum {
  IX_X = 0, IX_LNIN_G, IX_LNIN_B, IX_WRANK, IX_DWCONV, IX_WOUT,
  IX_GP_LNG, IX_GP_LNB, IX_GP_L1W, IX_GP_L1B, IX_GP_L2W, IX_GP_L2B,
  IX_CP_LNG, IX_CP_LNB, IX_CP_L1W, IX_CP_L1B, IX_CP_L2W, IX_CP_L2B,
  IX_GBLK0 = 18, IX_GBLK1 = 30, IX_CBLK0 = 42, IX_CBLK1 = 54,
  IX_FILM_W = 66, IX_FILM_B = 67,
  IX_BAT_LNG = 68, IX_BAT_LNB = 69, IX_BAT_W = 70, IX_BAT_B = 71,
  IX_GL_LNG = 72, IX_GL_LNB = 73, IX_GL_L1W = 74, IX_GL_L1B = 75, IX_GL_L2W = 76, IX_GL_L2B = 77,
  IX_SG_W = 78, IX_SG_B = 79, IX_ALPHA = 80,
  IX_SC_W = 81, IX_SC_B = 82,
  IX_RR_LNG = 83, IX_RR_LNB = 84, IX_RR_L1W = 85, IX_RR_L1B = 86, IX_RR_L2W = 87, IX_RR_L2B = 88,
  IX_GAIN = 89, IX_EMA = 90
};

struct BlkW { const unsigned short *wqkv, *wo, *ff1, *ff2; };

// One pre-LN transformer block (D in {64,768}), updates xb in place.
static void run_block(hipStream_t st, void* const* din, int bi, const BlkW& w, float* xb,
                      int D, int heads, int hid,
                      unsigned short* ln_bf, unsigned short* qkv_bf, float* scores,
                      unsigned short* probs_bf, unsigned short* vt_bf,
                      unsigned short* attno_bf, unsigned short* hid_bf) {
  const int M = 1024, Nt = 128, Bb = 8;
  int dh = D / heads;
  k_ln<<<dim3(M), dim3(256), 0, st>>>(xb, (const float*)din[bi + 0], (const float*)din[bi + 1], ln_bf, D, D);
  { GemmP p = G0(); p.A = ln_bf; p.B = w.wqkv; p.bias = (const float*)din[bi + 3];
    p.Cb = qkv_bf; p.M = M; p.N = 3 * D; p.K = D; p.lda = D; p.ldb = D; p.ldc = 3 * D;
    gemm_launch(st, p, 1); }
  { GemmP p = G0(); p.A = qkv_bf; p.B = qkv_bf + D; p.Cf = scores;           // Q @ K^T
    p.M = Nt; p.N = Nt; p.K = dh; p.lda = 3 * D; p.ldb = 3 * D; p.ldc = Nt;
    p.H = heads; p.sAb = (long)Nt * 3 * D; p.sAh = dh; p.sBb = (long)Nt * 3 * D; p.sBh = dh;
    p.sCb = (long)heads * Nt * Nt; p.sCh = (long)Nt * Nt; p.alpha = 1.f / sqrtf((float)dh);
    gemm_launch(st, p, Bb * heads); }
  k_softmax<<<dim3(Bb * heads * Nt), dim3(128), 0, st>>>(scores, probs_bf);
  {                                                                           // V^T
    long total = (long)Bb * heads * dh * Nt;
    k_vtrans<<<dim3((unsigned)((total + 255) / 256)), dim3(256), 0, st>>>(
        qkv_bf, vt_bf, D, dh, heads, total);
  }
  { GemmP p = G0(); p.A = probs_bf; p.B = vt_bf; p.Cb = attno_bf;            // P @ V
    p.M = Nt; p.N = dh; p.K = Nt; p.lda = Nt; p.ldb = Nt; p.ldc = D;
    p.H = heads; p.sAb = (long)heads * Nt * Nt; p.sAh = (long)Nt * Nt;
    p.sBb = (long)heads * dh * Nt; p.sBh = (long)dh * Nt;
    p.sCb = (long)Nt * D; p.sCh = dh;
    gemm_launch(st, p, Bb * heads); }
  { GemmP p = G0(); p.A = attno_bf; p.B = w.wo; p.bias = (const float*)din[bi + 5];
    p.resid = xb; p.Cf = xb; p.M = M; p.N = D; p.K = D; p.lda = D; p.ldb = D; p.ldc = D;
    gemm_launch(st, p, 1); }
  k_ln<<<dim3(M), dim3(256), 0, st>>>(xb, (const float*)din[bi + 6], (const float*)din[bi + 7], ln_bf, D, D);
  { GemmP p = G0(); p.A = ln_bf; p.B = w.ff1; p.bias = (const float*)din[bi + 9];
    p.Cb = hid_bf; p.M = M; p.N = hid; p.K = D; p.lda = D; p.ldb = D; p.ldc = hid; p.act = 1;
    gemm_launch(st, p, 1); }
  { GemmP p = G0(); p.A = hid_bf; p.B = w.ff2; p.bias = (const float*)din[bi + 11];
    p.resid = xb; p.Cf = xb; p.M = M; p.N = D; p.K = hid; p.lda = hid; p.ldb = hid; p.ldc = D;
    gemm_launch(st, p, 1); }
}

extern "C" void kernel_launch(void* const* d_in, const int* in_sizes, int n_in,
                              void* d_out, int out_size, void* d_ws, size_t ws_size,
                              hipStream_t stream) {
  (void)in_sizes; (void)out_size; (void)ws_size;
  if (n_in < 91) return;

  char* ws = (char*)d_ws;
  size_t off = 0;
  auto alloc = [&](size_t bytes) -> char* {
    char* p = ws + off; off += bytes; off = (off + 255) & ~(size_t)255; return p;
  };
  unsigned short* h_bf     = (unsigned short*)alloc(32768L * 768 * 2);
  float* rbuf              = (float*)alloc(32768L * 16 * 4);
  float* r2                = (float*)alloc(32768L * 16 * 4);
  float* rowscale          = (float*)alloc(32768L * 4);
  float* Ub                = (float*)alloc(32768L * 16 * 4);
  float* Vhb               = (float*)alloc(1024L * 256 * 4);
  float* Sb                = (float*)alloc(1024L * 16 * 4);
  float* gfeats            = (float*)alloc(1024L * 119 * 4);
  float* cfeats            = (float*)alloc(1024L * 51 * 4);
  unsigned short* ln_bf    = (unsigned short*)alloc(1024L * 1632 * 2);
  unsigned short* qkv_bf   = (unsigned short*)alloc(1024L * 2304 * 2);
  float* scores            = (float*)alloc(32L * 128 * 128 * 4);
  unsigned short* probs_bf = (unsigned short*)alloc(32L * 128 * 128 * 2);
  unsigned short* vt_bf    = (unsigned short*)alloc(32L * 192 * 128 * 2);
  unsigned short* attno_bf = (unsigned short*)alloc(1024L * 768 * 2);
  unsigned short* hid_bf   = (unsigned short*)alloc(1024L * 1536 * 2);
  float* geom              = (float*)alloc(1024L * 64 * 4);
  float* content           = (float*)alloc(1024L * 768 * 4);
  unsigned short* geom_bf  = (unsigned short*)alloc(1024L * 64 * 2);
  float* delta             = (float*)alloc(1024L * 16 * 4);
  float* scaleg            = (float*)alloc(1024L * 4);
  float* rf                = (float*)alloc(1024L * 16 * 4);
  float* rfcat             = (float*)alloc(1024L * 848 * 4);
  float* film              = (float*)alloc(1024L * 1536 * 4);
  float* cmod              = (float*)alloc(1024L * 768 * 4);
  float* batcat            = (float*)alloc(1024L * 832 * 4);
  float* gfeat             = (float*)alloc(8L * 1616 * 4);
  unsigned short* gmid_bf  = (unsigned short*)alloc(8L * 768 * 2);
  unsigned short* recon_bf = (unsigned short*)alloc(32768L * 32 * 2);

  // Weight conversion f32 -> zero-padded bf16 (L2-resident; one cheap pass)
  auto cvtw = [&](int idx, int N, int K) -> unsigned short* {
    int Kp = (K + 31) & ~31;
    unsigned short* dst = (unsigned short*)alloc((size_t)N * Kp * 2);
    long total = (long)N * Kp;
    k_wcvt<<<dim3((unsigned)((total + 255) / 256)), dim3(256), 0, stream>>>(
        (const float*)d_in[idx], dst, N, K, Kp);
    return dst;
  };
  unsigned short* W_rank = cvtw(IX_WRANK, 16, 768);
  unsigned short* W_out  = cvtw(IX_WOUT, 768, 16);      // Kp = 32
  unsigned short* W_gp1  = cvtw(IX_GP_L1W, 64, 119);    // Kp = 128
  unsigned short* W_gp2  = cvtw(IX_GP_L2W, 64, 64);
  unsigned short* W_cp1  = cvtw(IX_CP_L1W, 768, 51);    // Kp = 64
  unsigned short* W_cp2  = cvtw(IX_CP_L2W, 768, 768);
  BlkW blkw[4];
  const int blk_base[4] = { IX_GBLK0, IX_GBLK1, IX_CBLK0, IX_CBLK1 };
  const int blk_D[4]    = { 64, 64, 768, 768 };
  const int blk_hid[4]  = { 128, 128, 1536, 1536 };
  for (int i = 0; i < 4; ++i) {
    int bi = blk_base[i], D = blk_D[i], hid = blk_hid[i];
    blkw[i].wqkv = cvtw(bi + 2, 3 * D, D);
    blkw[i].wo   = cvtw(bi + 4, D, D);
    blkw[i].ff1  = cvtw(bi + 8, hid, D);
    blkw[i].ff2  = cvtw(bi + 10, D, hid);
  }
  unsigned short* W_film = cvtw(IX_FILM_W, 1536, 64);
  unsigned short* W_bat  = cvtw(IX_BAT_W, 768, 832);
  unsigned short* W_gl1  = cvtw(IX_GL_L1W, 768, 1616);  // Kp = 1632
  unsigned short* W_gl2  = cvtw(IX_GL_L2W, 768, 768);
  unsigned short* W_sg   = cvtw(IX_SG_W, 16, 64);
  unsigned short* W_sc   = cvtw(IX_SC_W, 1, 64);
  unsigned short* W_rr1  = cvtw(IX_RR_L1W, 32, 848);    // Kp = 864
  unsigned short* W_rr2  = cvtw(IX_RR_L2W, 16, 32);

  const float* x = (const float*)d_in[IX_X];
  float* y       = (float*)d_out;
  float* battery = y + 25165824;        // 8*4096*768
  float* gtok    = battery + 786432;    // 8*128*768

  // 1) token path: LN -> rank projection -> depthwise conv
  k_ln<<<dim3(32768), dim3(256), 0, stream>>>(x, (const float*)d_in[IX_LNIN_G],
                                              (const float*)d_in[IX_LNIN_B], h_bf, 768, 768);
  { GemmP p = G0(); p.A = h_bf; p.B = W_rank; p.Cf = rbuf;
    p.M = 32768; p.N = 16; p.K = 768; p.lda = 768; p.ldb = 768; p.ldc = 16;
    gemm_launch(stream, p, 1); }
  k_dwconv<<<dim3((32768 * 16 + 255) / 256), dim3(256), 0, stream>>>(
      rbuf, (const float*)d_in[IX_DWCONV], r2, 32768L * 16);

  // 2) per-group SVD + spectral features
  k_svd<<<dim3(1024), dim3(32), 0, stream>>>(r2, (const float*)d_in[IX_EMA],
                                             Ub, Vhb, Sb, rowscale, gfeats, cfeats);

  // 3) geom_proj / content_proj (LN -> lin -> GELU -> lin)
  k_ln<<<dim3(1024), dim3(256), 0, stream>>>(gfeats, (const float*)d_in[IX_GP_LNG],
                                             (const float*)d_in[IX_GP_LNB], ln_bf, 119, 128);
  { GemmP p = G0(); p.A = ln_bf; p.B = W_gp1; p.bias = (const float*)d_in[IX_GP_L1B];
    p.Cb = hid_bf; p.M = 1024; p.N = 64; p.K = 119; p.lda = 128; p.ldb = 128; p.ldc = 64; p.act = 1;
    gemm_launch(stream, p, 1); }
  { GemmP p = G0(); p.A = hid_bf; p.B = W_gp2; p.bias = (const float*)d_in[IX_GP_L2B];
    p.Cf = geom; p.M = 1024; p.N = 64; p.K = 64; p.lda = 64; p.ldb = 64; p.ldc = 64;
    gemm_launch(stream, p, 1); }
  k_ln<<<dim3(1024), dim3(256), 0, stream>>>(cfeats, (const float*)d_in[IX_CP_LNG],
                                             (const float*)d_in[IX_CP_LNB], ln_bf, 51, 64);
  { GemmP p = G0(); p.A = ln_bf; p.B = W_cp1; p.bias = (const float*)d_in[IX_CP_L1B];
    p.Cb = hid_bf; p.M = 1024; p.N = 768; p.K = 51; p.lda = 64; p.ldb = 64; p.ldc = 768; p.act = 1;
    gemm_launch(stream, p, 1); }
  { GemmP p = G0(); p.A = hid_bf; p.B = W_cp2; p.bias = (const float*)d_in[IX_CP_L2B];
    p.Cf = content; p.M = 1024; p.N = 768; p.K = 768; p.lda = 768; p.ldb = 768; p.ldc = 768;
    gemm_launch(stream, p, 1); }

  // 4) transformer stacks
  run_block(stream, d_in, IX_GBLK0, blkw[0], geom, 64, 4, 128, ln_bf, qkv_bf, scores, probs_bf, vt_bf, attno_bf, hid_bf);
  run_block(stream, d_in, IX_GBLK1, blkw[1], geom, 64, 4, 128, ln_bf, qkv_bf, scores, probs_bf, vt_bf, attno_bf, hid_bf);
  run_block(stream, d_in, IX_CBLK0, blkw[2], content, 768, 4, 1536, ln_bf, qkv_bf, scores, probs_bf, vt_bf, attno_bf, hid_bf);
  run_block(stream, d_in, IX_CBLK1, blkw[3], content, 768, 4, 1536, ln_bf, qkv_bf, scores, probs_bf, vt_bf, attno_bf, hid_bf);

  // 5) gates
  k_cvt<<<dim3((65536 + 255) / 256), dim3(256), 0, stream>>>(geom, geom_bf, 65536);
  { GemmP p = G0(); p.A = geom_bf; p.B = W_sg; p.bias = (const float*)d_in[IX_SG_B];
    p.Cf = delta; p.M = 1024; p.N = 16; p.K = 64; p.lda = 64; p.ldb = 64; p.ldc = 16; p.act = 2;
    gemm_launch(stream, p, 1); }
  { GemmP p = G0(); p.A = geom_bf; p.B = W_sc; p.bias = (const float*)d_in[IX_SC_B];
    p.Cf = scaleg; p.M = 1024; p.N = 1; p.K = 64; p.lda = 64; p.ldb = 64; p.ldc = 1; p.act = 2;
    gemm_launch(stream, p, 1); }

  // 6) row refinement MLP
  k_rfcat<<<dim3((int)((1024L * 848 + 255) / 256)), dim3(256), 0, stream>>>(Sb, content, geom, rfcat, 1024L * 848);
  k_ln<<<dim3(1024), dim3(256), 0, stream>>>(rfcat, (const float*)d_in[IX_RR_LNG],
                                             (const float*)d_in[IX_RR_LNB], ln_bf, 848, 864);
  { GemmP p = G0(); p.A = ln_bf; p.B = W_rr1; p.bias = (const float*)d_in[IX_RR_L1B];
    p.Cb = hid_bf; p.M = 1024; p.N = 32; p.K = 848; p.lda = 864; p.ldb = 864; p.ldc = 32; p.act = 1;
    gemm_launch(stream, p, 1); }
  { GemmP p = G0(); p.A = hid_bf; p.B = W_rr2; p.bias = (const float*)d_in[IX_RR_L2B];
    p.Cf = rf; p.M = 1024; p.N = 16; p.K = 32; p.lda = 32; p.ldb = 32; p.ldc = 16;
    gemm_launch(stream, p, 1); }

  // 7) reconstruction + y = x + gain * (recon @ w_out^T)
  k_recon<<<dim3(1024), dim3(32), 0, stream>>>(Ub, Vhb, Sb, rowscale, delta,
                                               (const float*)d_in[IX_ALPHA], scaleg, rf, recon_bf);
  { GemmP p = G0(); p.A = recon_bf; p.B = W_out;
    p.resid = x; p.Cf = y; p.alpha_ptr = (const float*)d_in[IX_GAIN];
    p.M = 32768; p.N = 768; p.K = 16; p.lda = 32; p.ldb = 32; p.ldc = 768;
    gemm_launch(stream, p, 1); }

  // 8) FiLM -> cmod -> battery
  { GemmP p = G0(); p.A = geom_bf; p.B = W_film; p.bias = (const float*)d_in[IX_FILM_B];
    p.Cf = film; p.M = 1024; p.N = 1536; p.K = 64; p.lda = 64; p.ldb = 64; p.ldc = 1536;
    gemm_launch(stream, p, 1); }
  k_cmod<<<dim3((int)((1024L * 768 + 255) / 256)), dim3(256), 0, stream>>>(film, content, cmod, batcat, 1024L * 768);
  k_batgeom<<<dim3((65536 + 255) / 256), dim3(256), 0, stream>>>(geom, batcat, 65536);
  k_ln<<<dim3(1024), dim3(256), 0, stream>>>(batcat, (const float*)d_in[IX_BAT_LNG],
                                             (const float*)d_in[IX_BAT_LNB], ln_bf, 832, 832);
  { GemmP p = G0(); p.A = ln_bf; p.B = W_bat; p.bias = (const float*)d_in[IX_BAT_B];
    p.Cf = battery; p.M = 1024; p.N = 768; p.K = 832; p.lda = 832; p.ldb = 832; p.ldc = 768;
    gemm_launch(stream, p, 1); }

  // 9) pooled global summary token
  k_gfeat<<<dim3(8), dim3(256), 0, stream>>>(cmod, battery, geom, Sb, gfeat);
  k_ln<<<dim3(8), dim3(256), 0, stream>>>(gfeat, (const float*)d_in[IX_GL_LNG],
                                          (const float*)d_in[IX_GL_LNB], ln_bf, 1616, 1632);
  { GemmP p = G0(); p.A = ln_bf; p.B = W_gl1; p.bias = (const float*)d_in[IX_GL_L1B];
    p.Cb = gmid_bf; p.M = 8; p.N = 768; p.K = 1616; p.lda = 1632; p.ldb = 1632; p.ldc = 768; p.act = 1;
    gemm_launch(stream, p, 1); }
  { GemmP p = G0(); p.A = gmid_bf; p.B = W_gl2; p.bias = (const float*)d_in[IX_GL_L2B];
    p.Cf = gtok; p.M = 8; p.N = 768; p.K = 768; p.lda = 768; p.ldb = 768; p.ldc = 768;
    gemm_launch(stream, p, 1); }
}
